// CrystalGraphConvNet_13391708030009
// MI455X (gfx1250) — matrix operand order repaired and checked by reading, not compiled, not run on hardware
//
#include <hip/hip_runtime.h>
#include <math.h>

// ---------------- types for CDNA5 WMMA / TDM ----------------
typedef __attribute__((ext_vector_type(16))) _Float16 v16h;
typedef __attribute__((ext_vector_type(8)))  _Float16 v8h;
typedef __attribute__((ext_vector_type(8)))  float    v8f;
typedef unsigned uint32x4 __attribute__((ext_vector_type(4)));
typedef int      int32x8  __attribute__((ext_vector_type(8)));
typedef int      int32x4  __attribute__((ext_vector_type(4)));

union AF16 { v16h v; v8h h[2]; };

// ---------------- problem constants ----------------
constexpr int NATOMS = 60000;
constexpr int NCRYS  = 600;
constexpr int APC    = 100;     // atoms per crystal
constexpr int MR     = 12;      // real neighbors
constexpr int ORIGF  = 92;
constexpr int NBRF   = 41;
constexpr int NBRP   = 48;      // padded nbr feature row (16B-aligned chunks)
constexpr int AFD    = 64;      // atom_fea_len
constexpr int HF     = 128;
constexpr int K1     = 192;     // padded 2*AF + NBR_F (169 -> 192)
constexpr int KE     = 96;      // padded ORIG_F (92 -> 96)
constexpr int NO     = 128;     // 2*AF
constexpr int NSTRIPS = NATOMS / 16;  // 3750 (exact)

__device__ __forceinline__ float softplusf(float x) {
    return (x > 20.f) ? x : log1pf(expf(x));
}
__device__ __forceinline__ float sigmoidf(float x) {
    return 1.f / (1.f + expf(-x));
}
__device__ __forceinline__ v8f wmma_f16(const AF16& a, const AF16& b, v8f c) {
    return __builtin_amdgcn_wmma_f32_16x16x32_f16(false, a.v, false, b.v,
                                                  (short)0, c, false, false);
}

// ---------------- weight packing: f32 [K][N] -> f16 transposed [N][Kpad] ----------------
__global__ void k_packW(const float* __restrict__ embW, const float* __restrict__ fcW,
                        const float* __restrict__ skipW,
                        _Float16* __restrict__ embWt, _Float16* __restrict__ fcWt,
                        _Float16* __restrict__ skipWt) {
    const int nEW = AFD * KE;          // 64*96
    const int nFW = 3 * NO * K1;       // 3*128*192
    const int nSW = 3 * AFD * AFD;     // 3*64*64
    int i = blockIdx.x * blockDim.x + threadIdx.x;
    if (i < nEW) {
        int n = i / KE, k = i % KE;
        embWt[i] = (k < ORIGF) ? (_Float16)embW[k * AFD + n] : (_Float16)0.f;
    } else if (i < nEW + nFW) {
        int j = i - nEW;
        int l = j / (NO * K1), rem = j % (NO * K1);
        int n = rem / K1, k = rem % K1;
        fcWt[j] = (k < 2 * AFD + NBRF) ? (_Float16)fcW[(l * (2 * AFD + NBRF) + k) * NO + n]
                                       : (_Float16)0.f;
    } else if (i < nEW + nFW + nSW) {
        int j = i - nEW - nFW;
        int l = j / (AFD * AFD), rem = j % (AFD * AFD);
        int n = rem / AFD, k = rem % AFD;
        skipWt[j] = (_Float16)skipW[(l * AFD + k) * AFD + n];
    }
}

// ---------------- feature packing: atom_fea -> f16 padded, nbr_fea -> f16 padded ----------------
__global__ void k_packF(const float* __restrict__ atom_fea, const float* __restrict__ nbr_fea,
                        _Float16* __restrict__ a16, _Float16* __restrict__ n16) {
    const long nA = (long)NATOMS * KE;
    const long total = nA + (long)NATOMS * MR * NBRP;
    for (long i = (long)blockIdx.x * blockDim.x + threadIdx.x; i < total;
         i += (long)gridDim.x * blockDim.x) {
        if (i < nA) {
            long r = i / KE; int k = (int)(i % KE);
            a16[i] = (k < ORIGF) ? (_Float16)atom_fea[r * ORIGF + k] : (_Float16)0.f;
        } else {
            long j = i - nA;
            long r = j / NBRP; int k = (int)(j % NBRP);
            n16[j] = (k < NBRF) ? (_Float16)nbr_fea[r * NBRF + k] : (_Float16)0.f;
        }
    }
}

__global__ void k_zero(float* p, int n) {
    int i = blockIdx.x * blockDim.x + threadIdx.x;
    if (i < n) p[i] = 0.f;
}

// ---------------- embedding GEMM: h = atom_fea @ embW + embB ----------------
__global__ __launch_bounds__(128) void k_embed(const _Float16* __restrict__ A,
                                               const _Float16* __restrict__ Bt,
                                               const float* __restrict__ embB,
                                               float* __restrict__ h32,
                                               _Float16* __restrict__ h16) {
    const int wave = threadIdx.x >> 5, lane = threadIdx.x & 31;
    const int strip = blockIdx.x * 4 + wave;
    if (strip >= NSTRIPS) return;
    const int l15 = lane & 15;
    const bool hi = (lane & 16) != 0;
    const int aoff = hi ? 8 : 0, boff = hi ? 8 : 0;
    const _Float16* arow = A + (long)(strip * 16 + l15) * KE;
    v8f c[4] = {};
#pragma unroll
    for (int kc = 0; kc < 3; ++kc) {
        AF16 a;
        a.h[0] = *(const v8h*)(arow + kc * 32 + aoff);
        a.h[1] = *(const v8h*)(arow + kc * 32 + 16 + aoff);
#pragma unroll
        for (int nt = 0; nt < 4; ++nt) {
            AF16 b;
            const _Float16* bp = Bt + (l15 + nt * 16) * KE + kc * 32 + boff;
            b.h[0] = *(const v8h*)bp;
            b.h[1] = *(const v8h*)(bp + 16);
            c[nt] = wmma_f16(a, b, c[nt]);
        }
    }
#pragma unroll
    for (int nt = 0; nt < 4; ++nt) {
        const int col = l15 + nt * 16;
        const float bias = embB[col];
#pragma unroll
        for (int v = 0; v < 8; ++v) {
            const long r = strip * 16 + v + (hi ? 8 : 0);
            const float val = c[nt][v] + bias;
            h32[r * AFD + col] = val;
            h16[r * AFD + col] = (_Float16)val;
        }
    }
}

// ---------------- conv GEMM (mode 0 = BN1 stats, mode 1 = apply) ----------------
// One wave == one atom (16 padded M-rows x 128 cols). Block = 4 waves.
// Weight tile (128x192 f16 = 48KB) staged ONCE per block into LDS via the
// Tensor Data Mover; A (tot rows) staged per-iteration with 16B chunk copies.
__global__ __launch_bounds__(128) void k_fc(const _Float16* __restrict__ h16,
                                            const _Float16* __restrict__ nbr16,
                                            const int* __restrict__ nbrIdx,
                                            const _Float16* __restrict__ Bt,
                                            const float* __restrict__ bnMean,
                                            const float* __restrict__ bnScale,
                                            const float* __restrict__ bnShift,
                                            float* __restrict__ statsOut,
                                            float* __restrict__ summedOut, int mode) {
    __shared__ alignas(16) _Float16 BtL[NO * K1];       // 49152 B
    __shared__ alignas(16) _Float16 At[4 * 16 * K1];    // 24576 B
    __shared__ float red[2 * NO];
    const int tid = threadIdx.x;
    const int wave = tid >> 5, lane = tid & 31;
    const int l15 = lane & 15;
    const bool hi = (lane & 16) != 0;
    const int aoff = hi ? 8 : 0, boff = hi ? 8 : 0;

    // ---- TDM: DMA the whole weight tile into LDS (wave 0 issues; EXEC-independent) ----
    if (tid < 32) {
        const unsigned ldsOff = (unsigned)(size_t)&BtL[0];   // low 32 bits = LDS byte addr
        const unsigned long long ga = (unsigned long long)(size_t)Bt;
        uint32x4 g0;
        g0[0] = 1u;                                   // count=1, user descriptor
        g0[1] = ldsOff;                               // lds_addr
        g0[2] = (unsigned)(ga & 0xFFFFFFFFu);         // global_addr[31:0]
        g0[3] = (unsigned)((ga >> 32) & 0x01FFFFFFu)  // global_addr[56:32]
              | (2u << 30);                           // type=2 ("image")
        int32x8 g1;
        g1[0] = (int)(2u << 16);          // wg_mask=0, data_size=2 (4 bytes)
        g1[1] = (int)(24576u << 16);      // tensor_dim0[15:0]=24576 dwords (128*192 f16)
        g1[2] = 0;                        // tensor_dim0 hi / tensor_dim1 lo
        g1[3] = (int)(24576u << 16);      // tile_dim0 = 24576 (flat 1-D tile)
        g1[4] = 0;                        // tile_dim1 / tile_dim2 unused
        g1[5] = 24576;                    // tensor_dim0_stride
        g1[6] = 0;
        g1[7] = 0;
        int32x4 gz4 = {0, 0, 0, 0};
        int32x8 gz8 = {0, 0, 0, 0, 0, 0, 0, 0};
#if defined(__has_builtin) && __has_builtin(__builtin_amdgcn_tensor_load_to_lds)
        __builtin_amdgcn_tensor_load_to_lds(g0, g1, gz4, gz4, gz8, 0);
        __builtin_amdgcn_s_wait_tensorcnt(0);
#else
        for (int i = lane; i < NO * K1 / 8; i += 32)
            *(v8h*)(&BtL[i * 8]) = *(const v8h*)(Bt + i * 8);
#endif
    }
    __syncthreads();

    float sAcc[8] = {0, 0, 0, 0, 0, 0, 0, 0};
    float qAcc[8] = {0, 0, 0, 0, 0, 0, 0, 0};

    float mF[4], sF[4], bF[4], mC[4], sC[4], bC[4];
    if (mode == 1) {
#pragma unroll
        for (int nt = 0; nt < 4; ++nt) {
            const int cf = l15 + nt * 16, cc = cf + AFD;
            mF[nt] = bnMean[cf]; sF[nt] = bnScale[cf]; bF[nt] = bnShift[cf];
            mC[nt] = bnMean[cc]; sC[nt] = bnScale[cc]; bC[nt] = bnShift[cc];
        }
    }

    for (int ab = blockIdx.x; ab < NATOMS / 4; ab += gridDim.x) {
        const int atomBase = ab * 4;
        __syncthreads();
        // build tot rows as 16-byte chunks: [h_self | h_nbr | nbr_fea(pad48) | 0]
        for (int t = tid; t < 4 * 16 * (K1 / 8); t += blockDim.x) {
            const int al = t / (16 * (K1 / 8));
            const int rem = t - al * (16 * (K1 / 8));
            const int m = rem / (K1 / 8);
            const int ch = rem - m * (K1 / 8);   // 0..23
            const int atom = atomBase + al;
            v8h val = {};
            if (m < MR) {
                if (ch < 8)
                    val = *(const v8h*)(h16 + (long)atom * AFD + ch * 8);
                else if (ch < 16)
                    val = *(const v8h*)(h16 + (long)nbrIdx[atom * MR + m] * AFD + (ch - 8) * 8);
                else if (ch < 22)
                    val = *(const v8h*)(nbr16 + ((long)atom * MR + m) * NBRP + (ch - 16) * 8);
            }
            *(v8h*)(At + ((al * 16 + m) * K1) + ch * 8) = val;
        }
        __syncthreads();

        const int atom = atomBase + wave;
        const _Float16* arow = &At[(wave * 16 + l15) * K1];
        v8f c[8] = {};
#pragma unroll
        for (int kc = 0; kc < 6; ++kc) {
            AF16 a;
            a.h[0] = *(const v8h*)(arow + kc * 32 + aoff);
            a.h[1] = *(const v8h*)(arow + kc * 32 + 16 + aoff);
#pragma unroll
            for (int nt = 0; nt < 8; ++nt) {
                AF16 b;
                const _Float16* bp = &BtL[(l15 + nt * 16) * K1 + kc * 32 + boff];
                b.h[0] = *(const v8h*)bp;
                b.h[1] = *(const v8h*)(bp + 16);
                c[nt] = wmma_f16(a, b, c[nt]);
            }
        }

        if (mode == 0) {
            // BN1 batch stats over the 12 REAL rows only (fcB absorbed analytically)
#pragma unroll
            for (int nt = 0; nt < 8; ++nt) {
                float s = 0.f, q = 0.f;
#pragma unroll
                for (int v = 0; v < 8; ++v) {
                    if (!hi || v < 4) {  // rows 0..11 real; 12..15 padding
                        const float g = c[nt][v];
                        s += g; q += g * g;
                    }
                }
                sAcc[nt] += s; qAcc[nt] += q;
            }
        } else {
            // normalize -> sigmoid(filter)*softplus(core) -> sum over m
#pragma unroll
            for (int nt = 0; nt < 4; ++nt) {
                float acc = 0.f;
#pragma unroll
                for (int v = 0; v < 8; ++v) {
                    if (!hi || v < 4) {
                        const float gf = (c[nt][v]     - mF[nt]) * sF[nt] + bF[nt];
                        const float gc = (c[nt + 4][v] - mC[nt]) * sC[nt] + bC[nt];
                        acc += sigmoidf(gf) * softplusf(gc);
                    }
                }
                acc += __shfl_xor(acc, 16, 32);  // rows 0-7 + rows 8-15
                if (!hi) summedOut[(long)atom * AFD + (l15 + nt * 16)] = acc;
            }
        }
    }

    if (mode == 0) {
        for (int i = tid; i < 2 * NO; i += blockDim.x) red[i] = 0.f;
        __syncthreads();
#pragma unroll
        for (int nt = 0; nt < 8; ++nt) {
            const int col = l15 + nt * 16;
            atomicAdd(&red[col], sAcc[nt]);
            atomicAdd(&red[NO + col], qAcc[nt]);
        }
        __syncthreads();
        for (int i = tid; i < 2 * NO; i += blockDim.x) atomicAdd(&statsOut[i], red[i]);
    }
}

// ---------------- BN finalize: stats -> mean/scale/shift ----------------
__global__ void k_bnfin(const float* __restrict__ stats, float cnt,
                        const float* __restrict__ gamma, const float* __restrict__ beta,
                        float* __restrict__ mean, float* __restrict__ scale,
                        float* __restrict__ shift, int C) {
    const int c = threadIdx.x;
    if (c >= C) return;
    const float s = stats[c], q = stats[C + c];
    const float mu = s / cnt;
    const float var = q / cnt - mu * mu;
    mean[c] = mu;
    scale[c] = gamma[c] * rsqrtf(var + 1e-5f);
    shift[c] = beta[c];
}

// ---------------- BN2 stats over summed [NATOMS][64] ----------------
__global__ void k_stats2(const float* __restrict__ summed, float* __restrict__ stats) {
    const int g = blockIdx.x * blockDim.x + threadIdx.x;  // 64 blocks * 256
    const int col = g & 63;
    const int r0 = g >> 6;  // 0..255
    float s = 0.f, q = 0.f;
    for (int r = r0; r < NATOMS; r += 256) {
        const float v = summed[(long)r * AFD + col];
        s += v; q += v * v;
    }
    atomicAdd(&stats[col], s);
    atomicAdd(&stats[AFD + col], q);
}

// ---------------- skip GEMM fused with BN2 + softplus ----------------
__global__ __launch_bounds__(128) void k_skip(const _Float16* __restrict__ Bt,
                                              const float* __restrict__ skipB,
                                              const float* __restrict__ summed,
                                              const float* __restrict__ m2,
                                              const float* __restrict__ s2,
                                              const float* __restrict__ b2,
                                              float* h32, _Float16* h16) {
    const int wave = threadIdx.x >> 5, lane = threadIdx.x & 31;
    const int strip = blockIdx.x * 4 + wave;
    if (strip >= NSTRIPS) return;
    const int l15 = lane & 15;
    const bool hi = (lane & 16) != 0;
    const int aoff = hi ? 8 : 0, boff = hi ? 8 : 0;
    const _Float16* arow = h16 + (long)(strip * 16 + l15) * AFD;
    v8f c[4] = {};
#pragma unroll
    for (int kc = 0; kc < 2; ++kc) {
        AF16 a;
        a.h[0] = *(const v8h*)(arow + kc * 32 + aoff);
        a.h[1] = *(const v8h*)(arow + kc * 32 + 16 + aoff);
#pragma unroll
        for (int nt = 0; nt < 4; ++nt) {
            AF16 b;
            const _Float16* bp = Bt + (l15 + nt * 16) * AFD + kc * 32 + boff;
            b.h[0] = *(const v8h*)bp;
            b.h[1] = *(const v8h*)(bp + 16);
            c[nt] = wmma_f16(a, b, c[nt]);
        }
    }
#pragma unroll
    for (int nt = 0; nt < 4; ++nt) {
        const int col = l15 + nt * 16;
        const float sb = skipB[col];
        const float mu = m2[col], sc = s2[col], sh = b2[col];
#pragma unroll
        for (int v = 0; v < 8; ++v) {
            const long r = strip * 16 + v + (hi ? 8 : 0);
            const float bn = (summed[r * AFD + col] - mu) * sc + sh;
            const float val = softplusf(c[nt][v] + sb + bn);
            h32[r * AFD + col] = val;
            h16[r * AFD + col] = (_Float16)val;
        }
    }
}

// ---------------- per-crystal mean ----------------
__global__ void k_segmean(const float* __restrict__ h32, float* __restrict__ crys) {
    const int b = blockIdx.x, c = threadIdx.x;
    float acc = 0.f;
    for (int a = 0; a < APC; ++a) acc += h32[((long)b * APC + a) * AFD + c];
    crys[b * AFD + c] = acc * (1.f / APC);
}

// ---------------- gate: ctx = sigmoid(crys @ gateW + gateB) ----------------
__global__ void k_gate(const float* __restrict__ crys, const float* __restrict__ W,
                       const float* __restrict__ B, float* __restrict__ ctx) {
    __shared__ float cm[AFD];
    const int b = blockIdx.x, j = threadIdx.x;
    cm[j] = crys[b * AFD + j];
    __syncthreads();
    float acc = B[j];
    for (int k = 0; k < AFD; ++k) acc += cm[k] * W[k * AFD + j];
    ctx[b * AFD + j] = sigmoidf(acc);
}

// ---------------- h += ctx[segment] ----------------
__global__ void k_addctx(float* h32, _Float16* h16, const float* __restrict__ ctx) {
    const long i = (long)blockIdx.x * blockDim.x + threadIdx.x;
    if (i >= (long)NATOMS * AFD) return;
    const int row = (int)(i >> 6), col = (int)(i & 63);
    const float v = h32[i] + ctx[(row / APC) * AFD + col];
    h32[i] = v;
    h16[i] = (_Float16)v;
}

// ---------------- head ----------------
__global__ void k_head(const float* __restrict__ crys, const float* __restrict__ cfW,
                       const float* __restrict__ cfB, const float* __restrict__ foW,
                       const float* __restrict__ foB, float* __restrict__ out) {
    __shared__ float cm[AFD];
    __shared__ float part[HF];
    const int b = blockIdx.x, j = threadIdx.x;
    if (j < AFD) cm[j] = crys[b * AFD + j];
    __syncthreads();
    float acc = cfB[j];
    for (int k = 0; k < AFD; ++k) acc += cm[k] * cfW[k * HF + j];
    part[j] = softplusf(acc) * foW[j];
    __syncthreads();
    for (int s = HF / 2; s > 0; s >>= 1) {
        if (j < s) part[j] += part[j + s];
        __syncthreads();
    }
    if (j == 0) out[b] = part[0] + foB[0];
}

// =====================================================================
extern "C" void kernel_launch(void* const* d_in, const int* in_sizes, int n_in,
                              void* d_out, int out_size, void* d_ws, size_t ws_size,
                              hipStream_t stream) {
    (void)in_sizes; (void)n_in; (void)out_size; (void)ws_size;
    const float* atom_fea = (const float*)d_in[0];
    const float* nbr_fea  = (const float*)d_in[1];
    const int*   nbrIdx   = (const int*)d_in[2];
    // d_in[3] segment_ids unused (contiguous blocks of 100)
    const float* embW  = (const float*)d_in[4];
    const float* embB  = (const float*)d_in[5];
    const float* fcW   = (const float*)d_in[6];
    // d_in[7] fcB unused: absorbed by BatchNorm (shifts mean, cancels in (x-mu))
    const float* bn1g  = (const float*)d_in[8];
    const float* bn1b  = (const float*)d_in[9];
    const float* bn2g  = (const float*)d_in[10];
    const float* bn2b  = (const float*)d_in[11];
    const float* skipW = (const float*)d_in[12];
    const float* skipB = (const float*)d_in[13];
    const float* gateW = (const float*)d_in[14];
    const float* gateB = (const float*)d_in[15];
    const float* cfW   = (const float*)d_in[16];
    const float* cfB   = (const float*)d_in[17];
    const float* foW   = (const float*)d_in[18];
    const float* foB   = (const float*)d_in[19];
    float* out = (float*)d_out;

    // ---- workspace carve (256B aligned) ----
    char* ws = (char*)d_ws;
    size_t off = 0;
    auto alloc = [&](size_t bytes) -> void* {
        void* p = ws + off;
        off += (bytes + 255) & ~(size_t)255;
        return p;
    };
    float* h32    = (float*)alloc((size_t)NATOMS * AFD * 4);
    float* summed = (float*)alloc((size_t)NATOMS * AFD * 4);
    float* crys   = (float*)alloc((size_t)NCRYS * AFD * 4);
    float* ctx    = (float*)alloc((size_t)NCRYS * AFD * 4);
    float* stats1 = (float*)alloc(2 * NO * 4);
    float* bn1p   = (float*)alloc(3 * NO * 4);   // mean | scale | shift
    float* stats2 = (float*)alloc(2 * AFD * 4);
    float* bn2p   = (float*)alloc(3 * AFD * 4);
    _Float16* h16    = (_Float16*)alloc((size_t)NATOMS * AFD * 2);
    _Float16* afea16 = (_Float16*)alloc((size_t)NATOMS * KE * 2);
    _Float16* nbr16  = (_Float16*)alloc((size_t)NATOMS * MR * NBRP * 2);
    _Float16* embWt  = (_Float16*)alloc((size_t)AFD * KE * 2);
    _Float16* fcWt   = (_Float16*)alloc((size_t)3 * NO * K1 * 2);
    _Float16* skipWt = (_Float16*)alloc((size_t)3 * AFD * AFD * 2);

    // ---- pack weights + features to f16 ----
    k_packW<<<(AFD * KE + 3 * NO * K1 + 3 * AFD * AFD + 255) / 256, 256, 0, stream>>>(
        embW, fcW, skipW, embWt, fcWt, skipWt);
    k_packF<<<2048, 256, 0, stream>>>(atom_fea, nbr_fea, afea16, nbr16);

    // ---- embedding ----
    k_embed<<<(NSTRIPS + 3) / 4, 128, 0, stream>>>(afea16, embWt, embB, h32, h16);

    // ---- conv layers ----
    for (int l = 0; l < 3; ++l) {
        const _Float16* BtFc = fcWt + (size_t)l * NO * K1;
        k_zero<<<1, 2 * NO, 0, stream>>>(stats1, 2 * NO);
        k_fc<<<1250, 128, 0, stream>>>(h16, nbr16, nbrIdx, BtFc,
                                       bn1p, bn1p + NO, bn1p + 2 * NO,
                                       stats1, summed, /*mode=*/0);
        k_bnfin<<<1, NO, 0, stream>>>(stats1, (float)(NATOMS * MR),
                                      bn1g + l * NO, bn1b + l * NO,
                                      bn1p, bn1p + NO, bn1p + 2 * NO, NO);
        k_fc<<<1250, 128, 0, stream>>>(h16, nbr16, nbrIdx, BtFc,
                                       bn1p, bn1p + NO, bn1p + 2 * NO,
                                       stats1, summed, /*mode=*/1);
        k_zero<<<1, 2 * AFD, 0, stream>>>(stats2, 2 * AFD);
        k_stats2<<<64, 256, 0, stream>>>(summed, stats2);
        k_bnfin<<<1, AFD, 0, stream>>>(stats2, (float)NATOMS,
                                       bn2g + l * AFD, bn2b + l * AFD,
                                       bn2p, bn2p + AFD, bn2p + 2 * AFD, AFD);
        k_skip<<<(NSTRIPS + 3) / 4, 128, 0, stream>>>(skipWt + (size_t)l * AFD * AFD,
                                                      skipB + l * AFD, summed,
                                                      bn2p, bn2p + AFD, bn2p + 2 * AFD,
                                                      h32, h16);
        k_segmean<<<NCRYS, AFD, 0, stream>>>(h32, crys);
        k_gate<<<NCRYS, AFD, 0, stream>>>(crys, gateW, gateB, ctx);
        k_addctx<<<(NATOMS * AFD + 255) / 256, 256, 0, stream>>>(h32, h16, ctx);
    }

    // ---- head ----
    k_segmean<<<NCRYS, AFD, 0, stream>>>(h32, crys);
    k_head<<<NCRYS, HF, 0, stream>>>(crys, cfW, cfB, foW, foB, out);
}
